// ComplexSSMLayer_62474594287862
// MI455X (gfx1250) — matrix operand order repaired and checked by reading, not compiled, not run on hardware
//
#include <hip/hip_runtime.h>
#include <hip/hip_bf16.h>
#include <math.h>

// ---------- types ----------
typedef __bf16 bf16_t;
typedef __attribute__((ext_vector_type(16))) __bf16 v16bf;
typedef __attribute__((ext_vector_type(8)))  __bf16 v8bf;
typedef __attribute__((ext_vector_type(4)))  __bf16 v4bf;
typedef __attribute__((ext_vector_type(8)))  float  v8f;

#define TOK   16384   // B*L
#define BATCH 4
#define SEQL  4096
#define DIM   1024
#define NST   128
#define K1    2048    // interleaved [xr0,xi0,xr1,xi1,...]
#define N1    384     // dt(128) + Bxr(128) + Bxi(128)
#define K2    256     // interleaved [hr0,hi0,hr1,hi1,...]
#define N2    2048    // yr(1024) + yi(1024)
#define NCH   32      // scan chunks
#define CL    128     // chunk length (NCH*CL == SEQL)

// workspace byte offsets
#define OFF_W1   ((size_t)0)                    // bf16 [24][64][32][16]   = 1,572,864 B
#define OFF_W2   ((size_t)1572864)              // bf16 [128][8][32][16]   = 1,048,576 B
#define OFF_A    ((size_t)2621440)              // f32  [TOK][128][2]      = 16 MiB
#define OFF_BX   ((size_t)19398656)             // f32  [TOK][128][2]      = 16 MiB
#define OFF_PS   ((size_t)36175872)             // f32  [B][32][128][4]    = 256 KiB
#define OFF_CR   ((size_t)36438016)             // f32  [B][32][128][2]    = 128 KiB
#define OFF_HBF  ((size_t)36700160)             // bf16 [TOK][256]         = 8 MiB

static __device__ __forceinline__ unsigned short bf_bits(float f) {
    unsigned u = __builtin_bit_cast(unsigned, f);
    unsigned r = u + 0x7FFFu + ((u >> 16) & 1u);     // round-to-nearest-even
    return (unsigned short)(r >> 16);
}
static __device__ __forceinline__ bf16_t f2bf(float f) {
    return __builtin_bit_cast(bf16_t, bf_bits(f));
}
static __device__ __forceinline__ v16bf cat16(v8bf lo, v8bf hi) {
    return __builtin_shufflevector(lo, hi, 0,1,2,3,4,5,6,7,8,9,10,11,12,13,14,15);
}

// ---------------------------------------------------------------------------
// Weight prep: swizzle combined [K1 x N1] matrix into per-lane WMMA B-fragment
// records: W1sw[nt][kci][lane][j], j = 16 contiguous bf16 per lane.
//   col = nt*16 + (lane&15);  k = kci*32 + (lane>>4)*16 + j
// K is the *interleaved* order: k = 2d + c  (c: 0 = multiplies xr, 1 = xi)
// ---------------------------------------------------------------------------
__global__ __launch_bounds__(256) void prep_w1(const float* __restrict__ dtW,
                                               const float* __restrict__ BWr,
                                               const float* __restrict__ BWi,
                                               bf16_t* __restrict__ W1sw) {
    int gid = blockIdx.x * 256 + threadIdx.x;           // < 24*64*32*16 = 786432
    int j    = gid & 15;
    int lane = (gid >> 4) & 31;
    int kci  = (gid >> 9) & 63;
    int nt   = gid >> 15;
    int col  = nt * 16 + (lane & 15);
    int k    = kci * 32 + (lane >> 4) * 16 + j;
    int d    = k >> 1;
    int c    = k & 1;                                   // 0: xr coeff, 1: xi coeff
    float v;
    if (col < 128) {                                    // dt: dt_W is [N][xr|xi]
        v = dtW[col * K1 + c * DIM + d];
    } else if (col < 256) {                             // Bxr = xr*BWr - xi*BWi
        int n = col - 128;
        v = c ? -BWi[n * DIM + d] : BWr[n * DIM + d];
    } else {                                            // Bxi = xr*BWi + xi*BWr
        int n = col - 256;
        v = c ?  BWr[n * DIM + d] : BWi[n * DIM + d];
    }
    W1sw[gid] = f2bf(v);
}

// W2 [K2 x N2]: k = 2n + c (c: 0 = hr coeff, 1 = hi); cols 0..1023 yr, 1024.. yi
__global__ __launch_bounds__(256) void prep_w2(const float* __restrict__ CWr,
                                               const float* __restrict__ CWi,
                                               bf16_t* __restrict__ W2sw) {
    int gid = blockIdx.x * 256 + threadIdx.x;           // < 128*8*32*16 = 524288
    int j    = gid & 15;
    int lane = (gid >> 4) & 31;
    int kci  = (gid >> 9) & 7;
    int nt   = gid >> 12;
    int col  = nt * 16 + (lane & 15);
    int k    = kci * 32 + (lane >> 4) * 16 + j;
    int n    = k >> 1;
    int c    = k & 1;
    int d    = col & (DIM - 1);
    bool im  = col >= DIM;
    float v;
    if (!im) v = c ? -CWi[d * NST + n] : CWr[d * NST + n];   // yr = hr*Cr - hi*Ci
    else     v = c ?  CWr[d * NST + n] : CWi[d * NST + n];   // yi = hr*Ci + hi*Cr
    W2sw[gid] = f2bf(v);
}

// ---------------------------------------------------------------------------
// GEMM-1 fused: per block = 16 tokens.  Interleaved K order makes the x tile a
// straight contiguous float4 copy + bf16 convert into LDS.  8 waves x 3
// N-tiles cover all 384 columns; K chunked 4x512.  Epilogue computes dt/A/Bx.
// ---------------------------------------------------------------------------
__global__ __launch_bounds__(256) void gemm1_fused(const float* __restrict__ x,
                                                   const float* __restrict__ lAr,
                                                   const float* __restrict__ lAi,
                                                   const float* __restrict__ dtb,
                                                   const float* __restrict__ dtbias,
                                                   const bf16_t* __restrict__ W1sw,
                                                   float* __restrict__ Abuf,
                                                   float* __restrict__ Bxbuf) {
    __shared__ __align__(16) bf16_t Ach[16][520];       // 512 + 8 pad (16B)
    __shared__ __align__(16) float  Gt[16][388];        // raw GEMM tile

    const int tid  = threadIdx.x;
    const int wid  = tid >> 5;
    const int lane = tid & 31;
    const int half = lane >> 4;
    const int row  = lane & 15;                         // M row / N col in tile
    const int t0   = blockIdx.x * 16;

    v8f acc[3];
    #pragma unroll
    for (int i = 0; i < 3; ++i) acc[i] = (v8f){0,0,0,0,0,0,0,0};

    for (int kcc = 0; kcc < 4; ++kcc) {
        __syncthreads();
        // stage contiguous float4 slice of x, convert to bf16 (packed b64 store)
        for (int idx = tid; idx < 2048; idx += 256) {
            int tl  = idx >> 7;                         // 128 float4 per row
            int q   = idx & 127;
            float4 v = ((const float4*)(x + (size_t)(t0 + tl) * K1 + kcc * 512))[q];
            v4bf o; o[0] = f2bf(v.x); o[1] = f2bf(v.y); o[2] = f2bf(v.z); o[3] = f2bf(v.w);
            *(v4bf*)&Ach[tl][q * 4] = o;
        }
        __syncthreads();
        // prefetch next chunk while WMMAs run
        if (kcc < 3)
            __builtin_prefetch(x + (size_t)(t0 + (tid & 15)) * K1 + (kcc + 1) * 512 + (tid >> 4) * 32, 0, 0);

        for (int ks = 0; ks < 16; ++ks) {
            const int kc = ks * 32;
            // A fragment: 16 contiguous k (k = 16*half + j), matching W1sw
            v8bf a_lo = *(const v8bf*)&Ach[row][kc + 16 * half];
            v8bf a_hi = *(const v8bf*)&Ach[row][kc + 16 * half + 8];
            v16bf afrag = cat16(a_lo, a_hi);
            const int kci = kcc * 16 + ks;              // 0..63
            #pragma unroll
            for (int nt = 0; nt < 3; ++nt) {
                const int ntile = wid * 3 + nt;
                v16bf bfrag = *(const v16bf*)&W1sw[(((size_t)(ntile * 64 + kci)) * 32 + lane) * 16];
                acc[nt] = __builtin_amdgcn_wmma_f32_16x16x32_bf16(
                    false, afrag, false, bfrag, (short)0, acc[nt], false, false);
            }
        }
    }

    // write raw results to LDS tile (C/D layout: VGPR r -> M = r + 8*half)
    #pragma unroll
    for (int nt = 0; nt < 3; ++nt) {
        const int ntile = wid * 3 + nt;
        #pragma unroll
        for (int r = 0; r < 8; ++r)
            Gt[r + 8 * half][ntile * 16 + row] = acc[nt][r];
    }
    __syncthreads();

    // pointwise epilogue -> A, Bx scan inputs
    for (int idx = tid; idx < 16 * NST; idx += 256) {
        int tl = idx >> 7, n = idx & 127;
        float z  = Gt[tl][n] + dtb[n] + dtbias[n];
        float dt = (z > 20.f) ? z : log1pf(expf(z));    // softplus
        float Amag = expf(-dt * expf(lAr[n]));
        float th = lAi[n];
        float2 a; a.x = Amag * cosf(th); a.y = Amag * sinf(th);
        float2 bx; bx.x = Gt[tl][128 + n] * dt; bx.y = Gt[tl][256 + n] * dt;
        size_t t = (size_t)(t0 + tl);
        ((float2*)Abuf)[t * NST + n]  = a;
        ((float2*)Bxbuf)[t * NST + n] = bx;
    }
}

// ---------------------------------------------------------------------------
// Scan pass 1: per (b, chunk, n) compute P = prod A, q = local scan tail
// ---------------------------------------------------------------------------
__global__ __launch_bounds__(256) void scan_pass1(const float* __restrict__ Abuf,
                                                  const float* __restrict__ Bxbuf,
                                                  float* __restrict__ Psum) {
    int gid = blockIdx.x * 256 + threadIdx.x;           // < B*NCH*NST = 16384
    int n = gid & 127, c = (gid >> 7) & 31, b = gid >> 12;
    float2 P; P.x = 1.f; P.y = 0.f;
    float2 q; q.x = 0.f; q.y = 0.f;
    size_t base = ((size_t)b * SEQL + c * CL) * NST + n;
    for (int i = 0; i < CL; ++i) {
        float2 a  = ((const float2*)Abuf)[base + (size_t)i * NST];
        float2 bx = ((const float2*)Bxbuf)[base + (size_t)i * NST];
        float2 nq; nq.x = a.x * q.x - a.y * q.y + bx.x;
                   nq.y = a.x * q.y + a.y * q.x + bx.y;  q = nq;
        float2 nP; nP.x = a.x * P.x - a.y * P.y;
                   nP.y = a.x * P.y + a.y * P.x;         P = nP;
    }
    float4 o; o.x = P.x; o.y = P.y; o.z = q.x; o.w = q.y;
    ((float4*)Psum)[gid] = o;
}

// Scan pass 2: sequential carry combine across chunks; emits h[:, -1]
__global__ __launch_bounds__(256) void scan_carry(const float* __restrict__ Psum,
                                                  float* __restrict__ Carry,
                                                  float* __restrict__ hlast) {
    int gid = blockIdx.x * 256 + threadIdx.x;
    if (gid >= BATCH * NST) return;
    int n = gid & 127, b = gid >> 7;
    float2 carry; carry.x = 0.f; carry.y = 0.f;
    for (int c = 0; c < NCH; ++c) {
        int idx = (b * NCH + c) * NST + n;
        ((float2*)Carry)[idx] = carry;                  // prefix BEFORE chunk c
        float4 pq = ((const float4*)Psum)[idx];
        float2 nc; nc.x = pq.x * carry.x - pq.y * carry.y + pq.z;
                   nc.y = pq.x * carry.y + pq.y * carry.x + pq.w;
        carry = nc;
    }
    hlast[(b * NST + n) * 2 + 0] = carry.x;
    hlast[(b * NST + n) * 2 + 1] = carry.y;
}

// Scan pass 3: recompute chunk scan with carry; emit h as bf16 interleaved
// hbf[t][2n] = hr, hbf[t][2n+1] = hi  (one packed 32-bit store per state)
__global__ __launch_bounds__(256) void scan_pass3(const float* __restrict__ Abuf,
                                                  const float* __restrict__ Bxbuf,
                                                  const float* __restrict__ Carry,
                                                  bf16_t* __restrict__ hbf) {
    int gid = blockIdx.x * 256 + threadIdx.x;
    int n = gid & 127, c = (gid >> 7) & 31, b = gid >> 12;
    float2 h = ((const float2*)Carry)[(b * NCH + c) * NST + n];
    size_t tok0 = (size_t)b * SEQL + c * CL;
    for (int i = 0; i < CL; ++i) {
        size_t t = tok0 + i;
        float2 a  = ((const float2*)Abuf)[t * NST + n];
        float2 bx = ((const float2*)Bxbuf)[t * NST + n];
        float2 nh; nh.x = a.x * h.x - a.y * h.y + bx.x;
                   nh.y = a.x * h.y + a.y * h.x + bx.y;  h = nh;
        unsigned pack = (unsigned)bf_bits(h.x) | ((unsigned)bf_bits(h.y) << 16);
        ((unsigned*)hbf)[t * NST + n] = pack;
    }
}

// ---------------------------------------------------------------------------
// GEMM-2 fused: [16 tokens x 256] x W2 -> 2048 cols; epilogue adds cmul(D, x).
// h tile staged via GLOBAL_LOAD_ASYNC_TO_LDS_B128 (ASYNCcnt), no VGPR round
// trip.  8 waves x 16 N-tiles; K = 8 WMMA steps, fully unrolled.
// ---------------------------------------------------------------------------
__global__ __launch_bounds__(256) void gemm2_fused(const bf16_t* __restrict__ hbf,
                                                   const bf16_t* __restrict__ W2sw,
                                                   const float* __restrict__ x,
                                                   const float* __restrict__ Dp,
                                                   float* __restrict__ y) {
    __shared__ __align__(16) bf16_t Hs[16][264];        // 256 + 8 pad
    const int tid  = threadIdx.x;
    const int wid  = tid >> 5;
    const int lane = tid & 31;
    const int half = lane >> 4;
    const int row  = lane & 15;
    const int t0   = blockIdx.x * 16;

    // async-copy h tile (bf16 already -- no conversion needed): 512 x 16B
    for (int idx = tid; idx < 512; idx += 256) {
        int tl = idx >> 5, kk = (idx & 31) * 8;
        unsigned lds_addr = (unsigned)(size_t)(void*)&Hs[tl][kk];
        unsigned long long gaddr =
            (unsigned long long)(size_t)(hbf + (size_t)(t0 + tl) * K2 + kk);
        asm volatile("global_load_async_to_lds_b128 %0, %1, off"
                     :: "v"(lds_addr), "v"(gaddr) : "memory");
    }
    asm volatile("s_wait_asynccnt 0x0" ::: "memory");
    __syncthreads();

    for (int nt = 0; nt < 16; ++nt) {
        const int ntile = wid * 16 + nt;                // 0..127
        v8f acc = (v8f){0,0,0,0,0,0,0,0};
        #pragma unroll
        for (int kci = 0; kci < 8; ++kci) {
            const int kc = kci * 32;
            v8bf a_lo = *(const v8bf*)&Hs[row][kc + 16 * half];
            v8bf a_hi = *(const v8bf*)&Hs[row][kc + 16 * half + 8];
            v16bf afrag = cat16(a_lo, a_hi);
            v16bf bfrag = *(const v16bf*)&W2sw[(((size_t)(ntile * 8 + kci)) * 32 + lane) * 16];
            acc = __builtin_amdgcn_wmma_f32_16x16x32_bf16(
                false, afrag, false, bfrag, (short)0, acc, false, false);
        }
        // epilogue: scatter to y[token][d][comp] with residual cmul(D, x)
        #pragma unroll
        for (int r = 0; r < 8; ++r) {
            int m   = r + 8 * half;
            size_t t = (size_t)(t0 + m);
            int cg  = ntile * 16 + row;
            int d   = cg & (DIM - 1);
            int comp = cg >> 10;
            float2 xx = ((const float2*)x)[t * DIM + d];
            float2 dd = ((const float2*)Dp)[d];
            float res = comp ? (dd.x * xx.y + dd.y * xx.x)
                             : (dd.x * xx.x - dd.y * xx.y);
            y[(t * DIM + d) * 2 + comp] = acc[r] + res;
        }
    }
}

// ---------------------------------------------------------------------------
extern "C" void kernel_launch(void* const* d_in, const int* in_sizes, int n_in,
                              void* d_out, int out_size, void* d_ws, size_t ws_size,
                              hipStream_t stream) {
    const float* x      = (const float*)d_in[0];
    const float* lAr    = (const float*)d_in[1];
    const float* lAi    = (const float*)d_in[2];
    const float* dtW    = (const float*)d_in[3];
    const float* dtb    = (const float*)d_in[4];
    const float* dtbias = (const float*)d_in[5];
    const float* BWr    = (const float*)d_in[6];
    const float* BWi    = (const float*)d_in[7];
    const float* CWr    = (const float*)d_in[8];
    const float* CWi    = (const float*)d_in[9];
    const float* Dp     = (const float*)d_in[10];

    char* ws = (char*)d_ws;
    bf16_t* W1sw  = (bf16_t*)(ws + OFF_W1);
    bf16_t* W2sw  = (bf16_t*)(ws + OFF_W2);
    float*  Abuf  = (float*)(ws + OFF_A);
    float*  Bxbuf = (float*)(ws + OFF_BX);
    float*  Psum  = (float*)(ws + OFF_PS);
    float*  Carry = (float*)(ws + OFF_CR);
    bf16_t* hbf   = (bf16_t*)(ws + OFF_HBF);

    float* y     = (float*)d_out;
    float* hlast = y + (size_t)TOK * DIM * 2;

    prep_w1<<<786432 / 256, 256, 0, stream>>>(dtW, BWr, BWi, W1sw);
    prep_w2<<<524288 / 256, 256, 0, stream>>>(CWr, CWi, W2sw);
    gemm1_fused<<<TOK / 16, 256, 0, stream>>>(x, lAr, lAi, dtb, dtbias, W1sw, Abuf, Bxbuf);
    scan_pass1<<<(BATCH * NCH * NST) / 256, 256, 0, stream>>>(Abuf, Bxbuf, Psum);
    scan_carry<<<2, 256, 0, stream>>>(Psum, Carry, hlast);
    scan_pass3<<<(BATCH * NCH * NST) / 256, 256, 0, stream>>>(Abuf, Bxbuf, Carry, hbf);
    gemm2_fused<<<TOK / 16, 256, 0, stream>>>(hbf, W2sw, x, Dp, y);
}
